// NextDiT_20521353740612
// MI455X (gfx1250) — matrix operand order repaired and checked by reading, not compiled, not run on hardware
//
#include <hip/hip_runtime.h>

#define SEQ   2048
#define DIMC  2048
#define HID   8192
#define NH    32
#define NKV   8
#define HD    64
#define QKV_N ((NH + 2 * NKV) * HD) /* 3072 */

typedef __bf16 v16bf __attribute__((ext_vector_type(16)));
typedef __bf16 v8bf  __attribute__((ext_vector_type(8)));
typedef __bf16 v4bf  __attribute__((ext_vector_type(4)));
typedef float  v8f   __attribute__((ext_vector_type(8)));

__device__ __forceinline__ v16bf mk16(v8bf lo, v8bf hi) {
  return __builtin_shufflevector(lo, hi, 0,1,2,3,4,5,6,7,8,9,10,11,12,13,14,15);
}
// A-operand (16x32 bf16): lane holds row (lane&15); K chunks {8g..8g+7} and {16+8g..16+8g+7}
__device__ __forceinline__ v16bf frag_a(const __bf16* rowp, int g) {
  return mk16(*(const v8bf*)(rowp + 8 * g), *(const v8bf*)(rowp + 16 + 8 * g));
}
// B-operand (32x16 bf16): lane holds col (lane&15); K chunks {8g..8g+7} and {16+8g..16+8g+7}
__device__ __forceinline__ v16bf frag_b(const __bf16* rowp, int g) {
  return mk16(*(const v8bf*)(rowp + 8 * g), *(const v8bf*)(rowp + 16 + 8 * g));
}
__device__ __forceinline__ v8f wmma_bf(v16bf a, v16bf b, v8f c) {
  return __builtin_amdgcn_wmma_f32_16x16x32_bf16(false, a, false, b, (short)0, c,
                                                 false, false);
}

// ---- CDNA5 async global->LDS DMA (ASYNCcnt-tracked, completes in order) ----
// Generic pointers to LDS carry the LDS offset in their low 32 bits
// (flat aperture).
__device__ __forceinline__ void async_ld16(unsigned lds_off, const void* gptr) {
  asm volatile("global_load_async_to_lds_b128 %0, %1, off"
               :
               : "v"(lds_off), "v"(gptr)
               : "memory");
}
__device__ __forceinline__ void async_wait0() {
  asm volatile("s_wait_asynccnt 0x0" ::: "memory");
}
__device__ __forceinline__ void async_wait4() {
  asm volatile("s_wait_asynccnt 0x4" ::: "memory");
}

__device__ __forceinline__ float block_sum_256(float v, float* red) {
  int t = threadIdx.x;
  red[t] = v;
  __syncthreads();
  for (int s = 128; s > 0; s >>= 1) {
    if (t < s) red[t] += red[t + s];
    __syncthreads();
  }
  float r = red[0];
  __syncthreads();
  return r;
}
__device__ __forceinline__ float silu_f(float x) { return x / (1.f + __expf(-x)); }

// -------------------- mod = silu(c) @ w_ada^T + b_ada (tanh on gates) ------
__global__ __launch_bounds__(256) void nd_mod(const float* __restrict__ c,
                                              const float* __restrict__ w_ada,
                                              const float* __restrict__ b_ada,
                                              float* __restrict__ mod) {
  __shared__ float red[256];
  int j = blockIdx.x;
  const float* wr = w_ada + (size_t)j * DIMC;
  float acc = 0.f;
  for (int i = threadIdx.x; i < DIMC; i += 256) acc += silu_f(c[i]) * wr[i];
  float tot = block_sum_256(acc, red);
  if (threadIdx.x == 0) {
    float v = tot + b_ada[j];
    int seg = j >> 11;  // 0:scale_msa 1:gate_msa 2:scale_mlp 3:gate_mlp
    if (seg == 1 || seg == 3) v = tanhf(v);
    mod[j] = v;
  }
}

// -------------------- h = rms_norm(x, w) * (1 + scale) -> bf16 -------------
__global__ __launch_bounds__(256) void nd_rmsnorm_scale(const float* __restrict__ X,
                                                        const float* __restrict__ w,
                                                        const float* __restrict__ scalev,
                                                        __bf16* __restrict__ out) {
  __shared__ float red[256];
  int r = blockIdx.x;
  const float* xr = X + (size_t)r * DIMC;
  float acc = 0.f;
  for (int i = threadIdx.x; i < DIMC; i += 256) { float v = xr[i]; acc += v * v; }
  float tot = block_sum_256(acc, red);
  float rs = rsqrtf(tot / (float)DIMC + 1e-5f);
  for (int i = threadIdx.x; i < DIMC; i += 256)
    out[(size_t)r * DIMC + i] = (__bf16)(xr[i] * rs * w[i] * (1.f + scalev[i]));
}

// -------------------- out = resid + gate * rms_norm(y, w) ------------------
__global__ __launch_bounds__(256) void nd_residual(const float* __restrict__ resid,
                                                   const float* __restrict__ y,
                                                   const float* __restrict__ w,
                                                   const float* __restrict__ gate,
                                                   float* __restrict__ out) {
  __shared__ float red[256];
  int r = blockIdx.x;
  const float* yr = y + (size_t)r * DIMC;
  float acc = 0.f;
  for (int i = threadIdx.x; i < DIMC; i += 256) { float v = yr[i]; acc += v * v; }
  float tot = block_sum_256(acc, red);
  float rs = rsqrtf(tot / (float)DIMC + 1e-5f);
  for (int i = threadIdx.x; i < DIMC; i += 256)
    out[(size_t)r * DIMC + i] =
        resid[(size_t)r * DIMC + i] + gate[i] * (yr[i] * rs * w[i]);
}

// -------------------- fp32 -> bf16 convert ---------------------------------
__global__ __launch_bounds__(256) void nd_cvt(const float* __restrict__ in,
                                              __bf16* __restrict__ out, int n) {
  int n4 = n >> 2;
  for (int i = blockIdx.x * 256 + threadIdx.x; i < n4; i += gridDim.x * 256) {
    float4 v = ((const float4*)in)[i];
    v4bf o;
    o[0] = (__bf16)v.x; o[1] = (__bf16)v.y; o[2] = (__bf16)v.z; o[3] = (__bf16)v.w;
    ((v4bf*)out)[i] = o;
  }
}

// -------------------- mid = bf16(silu(g1) * g3) ----------------------------
__global__ __launch_bounds__(256) void nd_silu_mul(const float* __restrict__ g1,
                                                   const float* __restrict__ g3,
                                                   __bf16* __restrict__ out, int n) {
  int n4 = n >> 2;
  for (int i = blockIdx.x * 256 + threadIdx.x; i < n4; i += gridDim.x * 256) {
    float4 a = ((const float4*)g1)[i];
    float4 b = ((const float4*)g3)[i];
    v4bf o;
    o[0] = (__bf16)(silu_f(a.x) * b.x);
    o[1] = (__bf16)(silu_f(a.y) * b.y);
    o[2] = (__bf16)(silu_f(a.z) * b.z);
    o[3] = (__bf16)(silu_f(a.w) * b.w);
    ((v4bf*)out)[i] = o;
  }
}

// -------------------- C[M,N] f32 = A[M,K]bf16 @ W[N,K]bf16^T ---------------
// 128x128 block tile, K step 32, double-buffered LDS filled by async DMA.
// 8 waves in 2(M) x 4(N); wave tile 64x32 (8 f32 accum tiles).
#define LDSTR 48
__global__ __launch_bounds__(256) void nd_gemm(const __bf16* __restrict__ A,
                                               const __bf16* __restrict__ W,
                                               float* __restrict__ C,
                                               int M, int N, int K) {
  __shared__ __bf16 As[2][128][LDSTR];
  __shared__ __bf16 Bs[2][128][LDSTR];
  const int tid = threadIdx.x;
  const int lane = tid & 31;
  const int wave = tid >> 5;
  const int wy = wave >> 2;  // 0..1
  const int wx = wave & 3;   // 0..3
  const int g = lane >> 4;
  const int mrow = lane & 15;
  const int ntiles = N >> 7;
  const int m0 = (blockIdx.x / ntiles) << 7;
  const int n0 = (blockIdx.x % ntiles) << 7;

  // staging: 1024 chunks of 8 bf16 per tile pair; this thread owns rows r0,r0+64
  const int r0 = tid >> 2;       // 0..63
  const int c0 = (tid & 3) * 8;  // 0,8,16,24
  const int r1 = r0 + 64;

  const __bf16* pa0 = A + (size_t)(m0 + r0) * K + c0;
  const __bf16* pa1 = A + (size_t)(m0 + r1) * K + c0;
  const __bf16* pb0 = W + (size_t)(n0 + r0) * K + c0;
  const __bf16* pb1 = W + (size_t)(n0 + r1) * K + c0;

  unsigned la0[2], la1[2], lb0[2], lb1[2];
#pragma unroll
  for (int b = 0; b < 2; ++b) {
    la0[b] = (unsigned)(size_t)&As[b][r0][c0];
    la1[b] = (unsigned)(size_t)&As[b][r1][c0];
    lb0[b] = (unsigned)(size_t)&Bs[b][r0][c0];
    lb1[b] = (unsigned)(size_t)&Bs[b][r1][c0];
  }

  v8f acc[4][2] = {};

  auto stage = [&](int b, int k0) {
    async_ld16(la0[b], pa0 + k0);
    async_ld16(la1[b], pa1 + k0);
    async_ld16(lb0[b], pb0 + k0);
    async_ld16(lb1[b], pb1 + k0);
  };
  auto compute = [&](int b) {
    v16bf af[4], bfr[2];
#pragma unroll
    for (int mt = 0; mt < 4; ++mt)
      af[mt] = frag_a(&As[b][wy * 64 + mt * 16 + mrow][0], g);
#pragma unroll
    for (int nt = 0; nt < 2; ++nt)
      bfr[nt] = frag_b(&Bs[b][wx * 32 + nt * 16 + mrow][0], g);
#pragma unroll
    for (int mt = 0; mt < 4; ++mt)
#pragma unroll
      for (int nt = 0; nt < 2; ++nt)
        acc[mt][nt] = wmma_bf(af[mt], bfr[nt], acc[mt][nt]);
  };

  const int nk = K >> 5;
  stage(0, 0);
  for (int t = 0; t < nk - 1; ++t) {
    stage((t + 1) & 1, (t + 1) * 32);  // DMA next tile while computing current
    async_wait4();                     // first 4 async ops (current tile) done
    __syncthreads();
    compute(t & 1);
    __syncthreads();                   // all waves done reading before overwrite
  }
  async_wait0();
  __syncthreads();
  compute((nk - 1) & 1);

  // store: C layout M = j + 8*g, N = lane&15
#pragma unroll
  for (int mt = 0; mt < 4; ++mt)
#pragma unroll
    for (int nt = 0; nt < 2; ++nt)
#pragma unroll
      for (int j = 0; j < 8; ++j) {
        int r = m0 + wy * 64 + mt * 16 + j + 8 * g;
        int cc = n0 + wx * 32 + nt * 16 + mrow;
        C[(size_t)r * N + cc] = acc[mt][nt][j];
      }
  (void)M;
}

// -------------------- per-head RMSNorm + RoPE + bf16 pack ------------------
__global__ __launch_bounds__(256) void nd_qkv_post(const float* __restrict__ QKV,
                                                   const float* __restrict__ qw,
                                                   const float* __restrict__ kw,
                                                   const float* __restrict__ fc,
                                                   const float* __restrict__ fs,
                                                   __bf16* __restrict__ QB,
                                                   __bf16* __restrict__ KB,
                                                   __bf16* __restrict__ VB) {
  int gidx = blockIdx.x * 8 + (threadIdx.x >> 5);
  int lane = threadIdx.x & 31;
  int s = gidx / 48;
  int h = gidx % 48;  // 0..31 q, 32..39 k, 40..47 v
  const float* src = QKV + (size_t)s * QKV_N + h * HD;
  float x0 = src[2 * lane], x1 = src[2 * lane + 1];
  if (h < 40) {
    float ss = x0 * x0 + x1 * x1;
    ss += __shfl_xor(ss, 1, 32);
    ss += __shfl_xor(ss, 2, 32);
    ss += __shfl_xor(ss, 4, 32);
    ss += __shfl_xor(ss, 8, 32);
    ss += __shfl_xor(ss, 16, 32);
    float rs = rsqrtf(ss / (float)HD + 1e-5f);
    const float* nw = (h < NH) ? qw : kw;
    x0 *= rs * nw[2 * lane];
    x1 *= rs * nw[2 * lane + 1];
    float cv = fc[(size_t)s * (HD / 2) + lane];
    float sv = fs[(size_t)s * (HD / 2) + lane];
    float y0 = x0 * cv - x1 * sv;
    float y1 = x0 * sv + x1 * cv;
    __bf16* d = (h < NH) ? (QB + ((size_t)h * SEQ + s) * HD)
                         : (KB + ((size_t)(h - NH) * SEQ + s) * HD);
    d[2 * lane] = (__bf16)y0;
    d[2 * lane + 1] = (__bf16)y1;
  } else {
    __bf16* d = VB + ((size_t)(h - 40) * SEQ + s) * HD;
    d[2 * lane] = (__bf16)x0;
    d[2 * lane + 1] = (__bf16)x1;
  }
}

// -------------------- flash attention (full mask), WMMA --------------------
// block: 1 head x 128 q rows; 8 waves each own 16 q rows; 32-key steps.
__global__ __launch_bounds__(256) void nd_attn(const __bf16* __restrict__ QB,
                                               const __bf16* __restrict__ KB,
                                               const __bf16* __restrict__ VB,
                                               __bf16* __restrict__ OUT) {
  __shared__ __bf16 Ks[32][80];        // [key][d]
  __shared__ __bf16 Vt[64][LDSTR];     // [d][key] (transposed)
  __shared__ __bf16 Ps[8][16][LDSTR];  // wave-private P tiles
  const int tid = threadIdx.x;
  const int lane = tid & 31;
  const int wave = tid >> 5;
  const int g = lane >> 4;
  const int mrow = lane & 15;
  const int h = blockIdx.x >> 4;  // 16 q-tiles per head
  const int qt = blockIdx.x & 15;
  const int hv = h >> 2;          // GQA group
  const int q0 = qt * 128 + wave * 16;

  v16bf qf[2];
  {
    const __bf16* qp = QB + ((size_t)h * SEQ + (q0 + mrow)) * HD;
    qf[0] = frag_a(qp, g);
    qf[1] = frag_a(qp + 32, g);
  }

  v8f acc[4] = {};
  float mrun[8], lrun[8];
#pragma unroll
  for (int j = 0; j < 8; ++j) { mrun[j] = -3.0e38f; lrun[j] = 0.f; }

  const int rS = tid >> 3;
  const int cS = (tid & 7) * 8;
  const unsigned ldsK = (unsigned)(size_t)&Ks[rS][cS];

  for (int kt = 0; kt < SEQ; kt += 32) {
    // K tile via async DMA; V tile via register transpose into LDS
    async_ld16(ldsK, &KB[((size_t)hv * SEQ + kt + rS) * HD + cS]);
    {
      v8bf vv = *(const v8bf*)&VB[((size_t)hv * SEQ + kt + rS) * HD + cS];
#pragma unroll
      for (int i = 0; i < 8; ++i) Vt[cS + i][rS] = vv[i];
    }
    async_wait0();
    __syncthreads();

    v8f sc[2] = {};
#pragma unroll
    for (int t = 0; t < 2; ++t)
#pragma unroll
      for (int c = 0; c < 2; ++c) {
        v16bf b = frag_b(&Ks[t * 16 + mrow][c * 32], g);
        sc[t] = wmma_bf(qf[c], b, sc[t]);
      }

#pragma unroll
    for (int j = 0; j < 8; ++j) {
      float s0 = sc[0][j] * 0.125f;  // 1/sqrt(64)
      float s1 = sc[1][j] * 0.125f;
      float vmax = fmaxf(s0, s1);
      vmax = fmaxf(vmax, __shfl_xor(vmax, 1, 32));
      vmax = fmaxf(vmax, __shfl_xor(vmax, 2, 32));
      vmax = fmaxf(vmax, __shfl_xor(vmax, 4, 32));
      vmax = fmaxf(vmax, __shfl_xor(vmax, 8, 32));
      float mnew = fmaxf(mrun[j], vmax);
      float corr = __expf(mrun[j] - mnew);
      float p0 = __expf(s0 - mnew);
      float p1 = __expf(s1 - mnew);
      float ps = p0 + p1;
      ps += __shfl_xor(ps, 1, 32);
      ps += __shfl_xor(ps, 2, 32);
      ps += __shfl_xor(ps, 4, 32);
      ps += __shfl_xor(ps, 8, 32);
      lrun[j] = lrun[j] * corr + ps;
      mrun[j] = mnew;
#pragma unroll
      for (int dt = 0; dt < 4; ++dt) acc[dt][j] *= corr;
      Ps[wave][j + 8 * g][mrow] = (__bf16)p0;
      Ps[wave][j + 8 * g][16 + mrow] = (__bf16)p1;
    }
    // in-wave LDS round-trip (DS ops are in-order within a wave)
    v16bf pf = frag_a(&Ps[wave][mrow][0], g);
#pragma unroll
    for (int dt = 0; dt < 4; ++dt) {
      v16bf b = frag_b(&Vt[dt * 16 + mrow][0], g);
      acc[dt] = wmma_bf(pf, b, acc[dt]);
    }
    __syncthreads();
  }

#pragma unroll
  for (int dt = 0; dt < 4; ++dt)
#pragma unroll
    for (int j = 0; j < 8; ++j) {
      int srow = q0 + j + 8 * g;
      int col = h * HD + dt * 16 + mrow;
      OUT[(size_t)srow * DIMC + col] = (__bf16)(acc[dt][j] / lrun[j]);
    }
}

// ===========================================================================
extern "C" void kernel_launch(void* const* d_in, const int* in_sizes, int n_in,
                              void* d_out, int out_size, void* d_ws, size_t ws_size,
                              hipStream_t stream) {
  (void)in_sizes; (void)n_in; (void)out_size; (void)ws_size;
  const float* x            = (const float*)d_in[0];
  // d_in[1] = x_mask: all ones -> full (non-causal) attention, unused
  const float* fc           = (const float*)d_in[2];
  const float* fs           = (const float*)d_in[3];
  const float* c            = (const float*)d_in[4];
  const float* w_qkv        = (const float*)d_in[5];
  const float* w_out        = (const float*)d_in[6];
  const float* q_norm_w     = (const float*)d_in[7];
  const float* k_norm_w     = (const float*)d_in[8];
  const float* attn_norm1_w = (const float*)d_in[9];
  const float* attn_norm2_w = (const float*)d_in[10];
  const float* ffn_norm1_w  = (const float*)d_in[11];
  const float* ffn_norm2_w  = (const float*)d_in[12];
  const float* w1           = (const float*)d_in[13];
  const float* w2           = (const float*)d_in[14];
  const float* w3           = (const float*)d_in[15];
  const float* w_ada        = (const float*)d_in[16];
  const float* b_ada        = (const float*)d_in[17];
  float* out = (float*)d_out;

  char* wsb = (char*)d_ws;
  size_t off = 0;
  auto take = [&](size_t bytes) -> char* {
    char* p = wsb + off;
    off = (off + bytes + 255) & ~(size_t)255;
    return p;
  };
  float*  MOD   = (float*)take((size_t)4 * DIMC * 4);
  float*  X1    = (float*)take((size_t)SEQ * DIMC * 4);
  __bf16* HBF   = (__bf16*)take((size_t)SEQ * DIMC * 2);
  __bf16* WB    = (__bf16*)take((size_t)HID * DIMC * 2);
  char*   BIG0  = take((size_t)SEQ * HID * 4);
  char*   BIG1  = take((size_t)SEQ * HID * 4);
  __bf16* MIDBF = (__bf16*)take((size_t)SEQ * HID * 2);

  float*  QKVF   = (float*)BIG0;                 // [SEQ, 3072] then reused as G1
  float*  G1     = (float*)BIG0;                 // [SEQ, HID]
  __bf16* QB     = (__bf16*)BIG1;                // [NH][SEQ][HD]
  __bf16* KB     = QB + (size_t)NH * SEQ * HD;   // [NKV][SEQ][HD]
  __bf16* VB     = KB + (size_t)NKV * SEQ * HD;  // [NKV][SEQ][HD]
  __bf16* ATTNBF = VB + (size_t)NKV * SEQ * HD;  // [SEQ, DIM] bf16
  float*  APROJ  = (float*)(ATTNBF + (size_t)SEQ * DIMC);  // [SEQ, DIM]
  float*  G3     = (float*)BIG1;                 // [SEQ, HID] (after APROJ dead)
  float*  FFNO   = (float*)BIG1;                 // [SEQ, DIM] (after G3 dead)

  // 1) adaLN modulation
  nd_mod<<<4 * DIMC, 256, 0, stream>>>(c, w_ada, b_ada, MOD);
  // 2) h = rms(x)*(1+scale_msa) -> bf16
  nd_rmsnorm_scale<<<SEQ, 256, 0, stream>>>(x, attn_norm1_w, MOD, HBF);
  // 3) qkv GEMM
  nd_cvt<<<4096, 256, 0, stream>>>(w_qkv, WB, QKV_N * DIMC);
  nd_gemm<<<(SEQ / 128) * (QKV_N / 128), 256, 0, stream>>>(HBF, WB, QKVF, SEQ, QKV_N, DIMC);
  // 4) per-head norm + rope + pack
  nd_qkv_post<<<SEQ * 48 / 8, 256, 0, stream>>>(QKVF, q_norm_w, k_norm_w, fc, fs, QB, KB, VB);
  // 5) attention
  nd_attn<<<NH * (SEQ / 128), 256, 0, stream>>>(QB, KB, VB, ATTNBF);
  // 6) output projection
  nd_cvt<<<4096, 256, 0, stream>>>(w_out, WB, DIMC * DIMC);
  nd_gemm<<<(SEQ / 128) * (DIMC / 128), 256, 0, stream>>>(ATTNBF, WB, APROJ, SEQ, DIMC, DIMC);
  // 7) x1 = x + tanh(gate_msa)*rms(attn)
  nd_residual<<<SEQ, 256, 0, stream>>>(x, APROJ, attn_norm2_w, MOD + DIMC, X1);
  // 8) h2 = rms(x1)*(1+scale_mlp) -> bf16
  nd_rmsnorm_scale<<<SEQ, 256, 0, stream>>>(X1, ffn_norm1_w, MOD + 2 * DIMC, HBF);
  // 9) FFN
  nd_cvt<<<4096, 256, 0, stream>>>(w1, WB, HID * DIMC);
  nd_gemm<<<(SEQ / 128) * (HID / 128), 256, 0, stream>>>(HBF, WB, G1, SEQ, HID, DIMC);
  nd_cvt<<<4096, 256, 0, stream>>>(w3, WB, HID * DIMC);
  nd_gemm<<<(SEQ / 128) * (HID / 128), 256, 0, stream>>>(HBF, WB, G3, SEQ, HID, DIMC);
  nd_silu_mul<<<8192, 256, 0, stream>>>(G1, G3, MIDBF, SEQ * HID);
  nd_cvt<<<4096, 256, 0, stream>>>(w2, WB, DIMC * HID);
  nd_gemm<<<(SEQ / 128) * (DIMC / 128), 256, 0, stream>>>(MIDBF, WB, FFNO, SEQ, DIMC, HID);
  // 10) out = x1 + tanh(gate_mlp)*rms(ffn)
  nd_residual<<<SEQ, 256, 0, stream>>>(X1, FFNO, ffn_norm2_w, MOD + 3 * DIMC, out);
}
